// TanhMultiheadAttention_7524782702743
// MI455X (gfx1250) — matrix operand order repaired and checked by reading, not compiled, not run on hardware
//
#include <hip/hip_runtime.h>

typedef __attribute__((ext_vector_type(16))) __bf16 v16bf;
typedef __attribute__((ext_vector_type(8)))  float  v8f;

union FragU { uint4 u[2]; v16bf v; };

// A-matrix fragment (16x32 bf16, MxK). Lane r=lane&15 holds row r; half h=lane>>4:
// VGPR0-3 <- K = h*8 .. h*8+7 ; VGPR4-7 <- K = 16+h*8 .. 16+h*8+7 (ISA 7.12.2)
__device__ __forceinline__ v16bf load_fragA(const __bf16* tile, int lde) {
  const int lane = threadIdx.x & 31;
  const int r = lane & 15;
  const int h = lane >> 4;
  const __bf16* p = tile + r * lde + h * 8;
  FragU f;
  f.u[0] = *(const uint4*)(p);
  f.u[1] = *(const uint4*)(p + 16);
  return f.v;
}

// B-matrix fragment (32x16 bf16, KxN), supplied as B^T rows (N x K row-major).
// Lane r holds column N=r; half h: VGPR0-3 <- K = h*8 .. h*8+7 ;
// VGPR4-7 <- K = 16+h*8 .. 16+h*8+7 (same K order as the A fragment).
__device__ __forceinline__ v16bf load_fragB(const __bf16* tile, int lde) {
  const int lane = threadIdx.x & 31;
  const int r = lane & 15;
  const int h = lane >> 4;
  const __bf16* p = tile + r * lde + h * 8;
  FragU f;
  f.u[0] = *(const uint4*)(p);
  f.u[1] = *(const uint4*)(p + 16);
  return f.v;
}

__device__ __forceinline__ void bf8_to_f32(uint4 u, float* f) {
  unsigned int w0 = u.x, w1 = u.y, w2 = u.z, w3 = u.w;
  f[0] = __uint_as_float(w0 << 16); f[1] = __uint_as_float(w0 & 0xffff0000u);
  f[2] = __uint_as_float(w1 << 16); f[3] = __uint_as_float(w1 & 0xffff0000u);
  f[4] = __uint_as_float(w2 << 16); f[5] = __uint_as_float(w2 & 0xffff0000u);
  f[6] = __uint_as_float(w3 << 16); f[7] = __uint_as_float(w3 & 0xffff0000u);
}

// One projection: y = x @ W^T + b for a 32-token tile, result (bf16) into LDS.
__device__ void project(const float* __restrict__ x32,   // [32][1024] fp32, contiguous
                        const __bf16* __restrict__ W,    // [1024 out][1024 in] bf16
                        const float* __restrict__ bias,  // [1024] fp32
                        __bf16* stage,                   // LDS [32][1024] bf16
                        __bf16* outLDS,                  // LDS [32][1024] bf16
                        int tid)
{
  // Stage activations fp32 -> bf16 (coalesced float4 loads).
  const float4* xv4 = (const float4*)x32;
  for (int e = tid; e < 8192; e += 256) {
    float4 f = xv4[e];
    __bf16* d = stage + e * 4;
    d[0] = (__bf16)f.x; d[1] = (__bf16)f.y; d[2] = (__bf16)f.z; d[3] = (__bf16)f.w;
  }
  __syncthreads();

  const int wave = tid >> 5;
  const int lane = tid & 31;
  const int r = lane & 15;
  const int h = lane >> 4;

  // 64 N-tiles grouped by 4; each job covers both M-tiles so B fragments are reused.
  for (int job = wave; job < 16; job += 8) {
    const int nt0 = job * 4;
    v8f acc0[4] = {};
    v8f acc1[4] = {};
    for (int kb = 0; kb < 32; ++kb) {
      v16bf a0 = load_fragA(stage + kb * 32, 1024);
      v16bf a1 = load_fragA(stage + 16 * 1024 + kb * 32, 1024);
#pragma unroll
      for (int nt = 0; nt < 4; ++nt) {
        v16bf b = load_fragB(W + (size_t)(nt0 + nt) * 16 * 1024 + kb * 32, 1024);
        acc0[nt] = __builtin_amdgcn_wmma_f32_16x16x32_bf16(
            false, a0, false, b, (short)0, acc0[nt], false, false);
        acc1[nt] = __builtin_amdgcn_wmma_f32_16x16x32_bf16(
            false, a1, false, b, (short)0, acc1[nt], false, false);
      }
    }
    // Epilogue: C/D layout -> lane holds (M = i + 8*h, N = r). Add bias, store bf16.
#pragma unroll
    for (int nt = 0; nt < 4; ++nt) {
      const int col = (nt0 + nt) * 16 + r;
      const float bv = bias[col];
#pragma unroll
      for (int i = 0; i < 8; ++i) {
        const int row = i + 8 * h;
        outLDS[row * 1024 + col]        = (__bf16)(acc0[nt][i] + bv);
        outLDS[(row + 16) * 1024 + col] = (__bf16)(acc1[nt][i] + bv);
      }
    }
  }
  __syncthreads();
}

__global__ void __launch_bounds__(256) convert_weights(
    const float* __restrict__ Wq, const float* __restrict__ Wk,
    const float* __restrict__ Wv, __bf16* __restrict__ out)
{
  int idx4 = blockIdx.x * 256 + threadIdx.x;          // 0..786431 (float4 units)
  const float* src = (idx4 < 262144) ? Wq : (idx4 < 524288) ? Wk : Wv;
  int local4 = idx4 & 262143;
  float4 f = ((const float4*)src)[local4];
  __bf16* d = out + (size_t)(idx4 >> 18) * 1048576 + (size_t)local4 * 4;
  d[0] = (__bf16)f.x; d[1] = (__bf16)f.y; d[2] = (__bf16)f.z; d[3] = (__bf16)f.w;
}

// LDS map (bytes): [0,64K) stage x-tile | [64K,128K) q | [128K,192K) k | [192K,256K) v
// After scores, q+k region is reused as fp32 ctx staging (exactly 128KB).
__global__ void __launch_bounds__(256) tanh_mha_fused(
    const float* __restrict__ xq, const float* __restrict__ xk,
    const float* __restrict__ xv,
    const __bf16* __restrict__ Wqb, const __bf16* __restrict__ Wkb,
    const __bf16* __restrict__ Wvb,
    const float* __restrict__ bq, const float* __restrict__ bk,
    const float* __restrict__ bv,
    float* __restrict__ ctx_out, float* __restrict__ probs_out)
{
  extern __shared__ char smem[];
  __bf16* stage = (__bf16*)(smem);
  __bf16* qL = (__bf16*)(smem + 65536);
  __bf16* kL = (__bf16*)(smem + 131072);
  __bf16* vL = (__bf16*)(smem + 196608);

  const int tid = threadIdx.x;
  const int tok0 = blockIdx.x * 32;

  project(xq + (size_t)tok0 * 1024, Wqb, bq, stage, qL, tid);
  project(xk + (size_t)tok0 * 1024, Wkb, bk, stage, kL, tid);
  project(xv + (size_t)tok0 * 1024, Wvb, bv, stage, vL, tid);

  // ---- per-token head-mixing attention (fp32 VALU; tiny FLOP count) ----
  const int local = tid >> 3;            // token within tile (0..31)
  const int sub = tid & 7;               // 8 threads per token -> 2 head-rows each
  const __bf16* qrow = qL + local * 1024;
  const __bf16* krow = kL + local * 1024;
  const __bf16* vrow = vL + local * 1024;
  const int n = tok0 + local;

  float pr[2][16];
#pragma unroll
  for (int hh = 0; hh < 2; ++hh) {
    const int hd = sub * 2 + hh;
    float s[16];
#pragma unroll
    for (int g = 0; g < 16; ++g) s[g] = 0.0f;
    for (int dc = 0; dc < 8; ++dc) {
      float qf[8];
      bf8_to_f32(*(const uint4*)(qrow + hd * 64 + dc * 8), qf);
#pragma unroll
      for (int g = 0; g < 16; ++g) {
        float kf[8];
        bf8_to_f32(*(const uint4*)(krow + g * 64 + dc * 8), kf);
#pragma unroll
        for (int e = 0; e < 8; ++e) s[g] = fmaf(qf[e], kf[e], s[g]);
      }
    }
#pragma unroll
    for (int g = 0; g < 16; ++g) {
      float p = tanhf(s[g] * 0.125f);    // 1/sqrt(64)
      pr[hh][g] = p;
      probs_out[(size_t)n * 256 + hd * 16 + g] = p;
    }
  }
  __syncthreads();                        // q,k no longer needed

  float* ctxS = (float*)(smem + 65536);   // fp32 [32][1024] overlay on q+k
#pragma unroll
  for (int hh = 0; hh < 2; ++hh) {
    const int hd = sub * 2 + hh;
    for (int dc = 0; dc < 8; ++dc) {
      float a[8];
#pragma unroll
      for (int e = 0; e < 8; ++e) a[e] = 0.0f;
#pragma unroll
      for (int g = 0; g < 16; ++g) {
        float vf[8];
        bf8_to_f32(*(const uint4*)(vrow + g * 64 + dc * 8), vf);
#pragma unroll
        for (int e = 0; e < 8; ++e) a[e] = fmaf(pr[hh][g], vf[e], a[e]);
      }
#pragma unroll
      for (int e = 0; e < 8; ++e)        // transposed layout: out[n, d*16 + h]
        ctxS[local * 1024 + (dc * 8 + e) * 16 + hd] = a[e];
    }
  }
  __syncthreads();

  // Coalesced writeout of the 32x1024 ctx tile.
  float4* dst = (float4*)(ctx_out + (size_t)tok0 * 1024);
  const float4* srcv = (const float4*)ctxS;
  for (int e = tid; e < 8192; e += 256) dst[e] = srcv[e];
}

extern "C" void kernel_launch(void* const* d_in, const int* in_sizes, int n_in,
                              void* d_out, int out_size, void* d_ws, size_t ws_size,
                              hipStream_t stream) {
  (void)in_sizes; (void)n_in; (void)out_size; (void)ws_size;
  const float* xq = (const float*)d_in[0];
  const float* xk = (const float*)d_in[1];
  const float* xv = (const float*)d_in[2];
  const float* Wq = (const float*)d_in[3];
  const float* bq = (const float*)d_in[4];
  const float* Wk = (const float*)d_in[5];
  const float* bk = (const float*)d_in[6];
  const float* Wv = (const float*)d_in[7];
  const float* bv = (const float*)d_in[8];

  __bf16* wb = (__bf16*)d_ws;            // 3 x 1M bf16 = 6 MB scratch
  convert_weights<<<3072, 256, 0, stream>>>(Wq, Wk, Wv, wb);

  float* ctx = (float*)d_out;
  float* probs = ctx + (size_t)32768 * 1024;

  (void)hipFuncSetAttribute((const void*)tanh_mha_fused,
                            hipFuncAttributeMaxDynamicSharedMemorySize, 262144);
  tanh_mha_fused<<<1024, 256, 262144, stream>>>(
      xq, xk, xv, wb, wb + 1048576, wb + 2097152, bq, bk, bv, ctx, probs);
}
